// SegCaps_9852654977073
// MI455X (gfx1250) — matrix operand order repaired and checked by reading, not compiled, not run on hardware
//
#include <hip/hip_runtime.h>
#include <hip/hip_bf16.h>

typedef __attribute__((ext_vector_type(16))) _Float16 v16h;
typedef __attribute__((ext_vector_type(8)))  _Float16 v8h;
typedef __attribute__((ext_vector_type(8)))  float    v8f;
typedef __attribute__((ext_vector_type(4)))  int      v4i;
typedef __attribute__((ext_vector_type(8)))  int      v8i;

#define NIMG 4

static inline int cdiv_host(int a, int b) { return (a + b - 1) / b; }
__device__ __forceinline__ int imin_d(int a, int b) { return a < b ? a : b; }

// ---------------------------------------------------------------------------
// Implicit-GEMM conv / dilated-deconv via WMMA f32<-f16 16x16x32.
// One wave computes a 32(pixel) x 16*NT(out-channel) tile: 2xNT WMMAs / chunk.
// K layout: k = tap*CinPad + cin, CinPad = 1<<cshift.
// BIGC (cshift>=5): a 32-wide K chunk lies inside ONE tap -> tap decode (ky,kx,
//   ih,iw,bounds,row base) is hoisted into persistent state updated only when
//   kk crosses a CinPad boundary (uniform scalar branch). Steady-state chunk =
//   loads + cndmask + WMMA only; VALU co-executes under the XDL matrix pipe.
// !BIGC (conv1/w1a, tiny layers): generic per-chunk decode.
// Fully predicated (no exec branches); double-buffered so next chunk's loads
// overlap current chunk's WMMAs.
// Input: NHWC f16, per-pixel stride Cstride, channel offset cin_off.
// Weights: wpk[cout][KK] f16 (transposed, zero-padded rows).
// Output: raw conv result f16 [n][OH*OW][Cout].
// ---------------------------------------------------------------------------
template<int NT, bool DEC, bool BIGC>
__global__ __launch_bounds__(32) void wmma_conv_kernel(
    const _Float16* __restrict__ in, const _Float16* __restrict__ wpk,
    _Float16* __restrict__ out,
    int Cin, int cshift, int Cstride, int cin_off,
    int Hin, int Win, int OH, int OW, int Cout, int stride)
{
  const int lane = threadIdx.x & 31;
  const int hi   = lane >> 4;
  const int ln   = lane & 15;
  const int m_base = blockIdx.x * 32;
  const int n_base = blockIdx.y * (16 * NT);
  const int nimg   = blockIdx.z;
  const int P = OH * OW;

  int oh[2], ow[2];
#pragma unroll
  for (int mt = 0; mt < 2; ++mt) {
    int p = m_base + mt * 16 + ln;
    oh[mt] = p / OW; ow[mt] = p % OW;
  }

  const _Float16* inp = in + (size_t)nimg * Hin * Win * Cstride;
  const int cmask = (1 << cshift) - 1;
  const int KK = ((25 << cshift) + 31) & ~31;
  const _Float16* bb[NT];
#pragma unroll
  for (int nt = 0; nt < NT; ++nt)
    bb[nt] = wpk + (size_t)(n_base + nt * 16 + ln) * KK + (hi << 4);

  // persistent per-tap state (BIGC path)
  int rowOff[2] = {0, 0};
  int okT[2]    = {0, 0};

  auto tap_update = [&](int kk) {
    int tap = kk >> cshift;                 // uniform
    int ky = tap / 5, kx = tap - ky * 5;
#pragma unroll
    for (int mt = 0; mt < 2; ++mt) {
      int ih, iw, ok;
      if (!DEC) {
        ih = oh[mt] * stride + ky - 2;
        iw = ow[mt] * stride + kx - 2;
        ok = ((unsigned)ih < (unsigned)Hin) & ((unsigned)iw < (unsigned)Win);
      } else {  // lhs_dilation=2, pad(2,3): tap contributes only on parity match
        int ny = oh[mt] + ky - 2, nx = ow[mt] + kx - 2;
        ih = ny >> 1; iw = nx >> 1;
        ok = (ny >= 0) & (nx >= 0) & !(ny & 1) & !(nx & 1) & (ih < Hin) & (iw < Win);
      }
      int ihc = ok ? ih : 0, iwc = ok ? iw : 0;
      rowOff[mt] = (ihc * Win + iwc) * Cstride + cin_off;
      okT[mt] = ok;
    }
  };

  // issue all loads for one 32-wide K chunk (predicated, branch-free per lane)
  auto load_chunk = [&](int kk, v4i (&Ar)[2][2], int (&Aok)[2][2], v8i (&Br)[NT]) {
#pragma unroll
    for (int nt = 0; nt < NT; ++nt)
      Br[nt] = *(const v8i*)(bb[nt] + kk);          // 2x global_load_b128
    if (BIGC) {
      if ((kk & cmask) == 0) tap_update(kk);        // uniform branch, rare
#pragma unroll
      for (int g2 = 0; g2 < 2; ++g2) {
        int cin = (kk & cmask) + (g2 << 3) + (hi << 4);
        int cv  = cin < Cin;                         // only != CinPad for w10
        int cinc = cv ? cin : 0;
#pragma unroll
        for (int mt = 0; mt < 2; ++mt) {
          Aok[g2][mt] = okT[mt] & cv;
          Ar[g2][mt] = *(const v4i*)(inp + rowOff[mt] + cinc);
        }
      }
    } else {
#pragma unroll
      for (int g2 = 0; g2 < 2; ++g2) {
        int k0  = kk + (g2 << 3) + (hi << 4);
        int cin = k0 & cmask;
        int tap = k0 >> cshift;
        int ok0 = (tap < 25) & (cin < Cin);
        int tapc = ok0 ? tap : 0;
        int cinc = ok0 ? cin : 0;
        int ky = tapc / 5, kx = tapc - ky * 5;
#pragma unroll
        for (int mt = 0; mt < 2; ++mt) {
          int ih = oh[mt] * stride + ky - 2;
          int iw = ow[mt] * stride + kx - 2;
          int ok = ok0 & ((unsigned)ih < (unsigned)Hin) & ((unsigned)iw < (unsigned)Win);
          int ihc = ok ? ih : 0, iwc = ok ? iw : 0;
          Aok[g2][mt] = ok;
          Ar[g2][mt] = *(const v4i*)(inp + (ihc * Win + iwc) * Cstride + cin_off + cinc);
        }
      }
    }
  };

  v8f c[2][NT];
#pragma unroll
  for (int mt = 0; mt < 2; ++mt)
#pragma unroll
    for (int nt = 0; nt < NT; ++nt)
#pragma unroll
      for (int e = 0; e < 8; ++e) c[mt][nt][e] = 0.f;

  auto consume = [&](v4i (&Ar)[2][2], int (&Aok)[2][2], v8i (&Br)[NT]) {
    const v4i vz = {0, 0, 0, 0};
    v16h a[2];
#pragma unroll
    for (int mt = 0; mt < 2; ++mt) {
      v4i r0 = Aok[0][mt] ? Ar[0][mt] : vz;        // v_cndmask x4
      v4i r1 = Aok[1][mt] ? Ar[1][mt] : vz;
      v8i ai;
#pragma unroll
      for (int j = 0; j < 4; ++j) { ai[j] = r0[j]; ai[4 + j] = r1[j]; }
      a[mt] = __builtin_bit_cast(v16h, ai);
    }
#pragma unroll
    for (int nt = 0; nt < NT; ++nt) {
      v16h b = __builtin_bit_cast(v16h, Br[nt]);
#pragma unroll
      for (int mt = 0; mt < 2; ++mt)
        c[mt][nt] = __builtin_amdgcn_wmma_f32_16x16x32_f16(
            false, a[mt], false, b, (short)0, c[mt][nt], false, false);
    }
  };

  // ---- double-buffered main loop ----
  v4i Ar0[2][2], Ar1[2][2];
  int Aok0[2][2], Aok1[2][2];
  v8i Br0[NT], Br1[NT];

  load_chunk(0, Ar0, Aok0, Br0);
  int kk = 0;
  for (;;) {
    if (kk + 32 < KK) load_chunk(kk + 32, Ar1, Aok1, Br1);
    consume(Ar0, Aok0, Br0);
    kk += 32;
    if (kk >= KK) break;
    if (kk + 32 < KK) load_chunk(kk + 32, Ar0, Aok0, Br0);
    consume(Ar1, Aok1, Br1);
    kk += 32;
    if (kk >= KK) break;
  }

  // C layout: VGPR v -> row m = v + 8*hi, lane -> col n
#pragma unroll
  for (int mt = 0; mt < 2; ++mt)
#pragma unroll
    for (int nt = 0; nt < NT; ++nt) {
      _Float16* op = out + ((size_t)nimg * P + m_base + mt * 16) * Cout + n_base + nt * 16;
#pragma unroll
      for (int v = 0; v < 8; ++v)
        op[(size_t)(v + (hi << 3)) * Cout + ln] = (_Float16)c[mt][nt][v];
    }
}

// ---------------------------------------------------------------------------
// Weight packer: reference layout -> wpk[cout][KK] f16, K = tap*CinPad + cin.
// caps conv  : w[t0][cout][z][ky][kx]
// caps deconv: w[t0][z][cout][ky][kx], spatially flipped (transposed conv)
// ---------------------------------------------------------------------------
__global__ void pack_w_kernel(const float* __restrict__ w, _Float16* __restrict__ wpk,
                              int t0, int z_in, int Cout, int cshift, int KK, int is_deconv)
{
  int idx = blockIdx.x * blockDim.x + threadIdx.x;
  if (idx >= Cout * KK) return;
  int n = idx / KK, k = idx - n * KK;
  int cmask = (1 << cshift) - 1;
  int cin = k & cmask, tap = k >> cshift;
  float val = 0.f;
  if (tap < 25 && cin < t0 * z_in) {
    int t = cin / z_in, z = cin - t * z_in;
    int ky = tap / 5, kx = tap - ky * 5;
    size_t src;
    if (!is_deconv)
      src = ((((size_t)t * Cout + n) * z_in + z) * 5 + ky) * 5 + kx;
    else
      src = ((((size_t)t * z_in + z) * Cout + n) * 5 + (4 - ky)) * 5 + (4 - kx);
    val = w[src];
  }
  wpk[idx] = (_Float16)val;
}

// NCHW f32 -> NHWC f16, channel-padded to 8 (zeros in ch 3..7) for conv1
__global__ void pack_x_kernel(const float* __restrict__ x, _Float16* __restrict__ xh,
                              int H, int W)
{
  int idx = blockIdx.x * blockDim.x + threadIdx.x;
  int total = NIMG * H * W * 8;
  if (idx >= total) return;
  int c = idx & 7; int t = idx >> 3;
  int w = t % W; t /= W;
  int h = t % H; int n = t / H;
  float v = (c < 3) ? x[(((size_t)n * 3 + c) * H + h) * W + w] : 0.f;
  xh[idx] = (_Float16)v;
}

// identity "routing" for conv1: strided f16 copy into a concat buffer
__global__ void cast_copy_kernel(const _Float16* __restrict__ src, _Float16* __restrict__ dst,
                                 int Cout, int cout_off, int Cstride, int totalP)
{
  int idx = blockIdx.x * blockDim.x + threadIdx.x;
  if (idx >= totalP * Cout) return;
  int c = idx % Cout; int p = idx / Cout;
  dst[(size_t)p * Cstride + cout_off + c] = src[idx];
}

// routing: uh_sum (+ bias summed over t0), scale by r=1/(t1*cnt), squash per z1 group
__global__ void route_kernel(const _Float16* __restrict__ conv,
                             const float* __restrict__ bias, int t0b,
                             _Float16* __restrict__ out, int cout_off, int Cstride,
                             int P, int OW, int OH, int Cout, int t1, int z1)
{
  int idx = blockIdx.x * blockDim.x + threadIdx.x;
  if (idx >= NIMG * P * t1) return;
  int g = idx % t1; int rem = idx / t1;
  int p = rem % P;  int n = rem / P;
  int oh = p / OW, ow = p % OW;
  int cy = imin_d(oh, 2) + imin_d(OH - 1 - oh, 2) + 1;
  int cx = imin_d(ow, 2) + imin_d(OW - 1 - ow, 2) + 1;
  float r = 1.f / ((float)t1 * (float)(cy * cx));
  const _Float16* cp = conv + ((size_t)n * P + p) * Cout + g * z1;

  float n2 = 0.f;
  for (int z = 0; z < z1; ++z) {
    float b = 0.f;
    if (bias) for (int t = 0; t < t0b; ++t) b += bias[t * Cout + g * z1 + z];
    float v = r * ((float)cp[z] + b);
    n2 += v * v;
  }
  float s = (n2 / (1.f + n2)) / sqrtf(n2 + 1e-9f);
  _Float16* op = out + ((size_t)n * P + p) * Cstride + cout_off + g * z1;
  for (int z = 0; z < z1; ++z) {
    float b = 0.f;
    if (bias) for (int t = 0; t < t0b; ++t) b += bias[t * Cout + g * z1 + z];
    float v = r * ((float)cp[z] + b);
    op[z] = (_Float16)(v * s);
  }
}

// final layer: routing (t1=1,z1=16) fused with channel-norm output
__global__ void final_kernel(const _Float16* __restrict__ conv, float* __restrict__ out,
                             int P, int OW, int OH)
{
  int idx = blockIdx.x * blockDim.x + threadIdx.x;
  if (idx >= NIMG * P) return;
  int p = idx % P; int n = idx / P;
  int oh = p / OW, ow = p % OW;
  int cy = imin_d(oh, 2) + imin_d(OH - 1 - oh, 2) + 1;
  int cx = imin_d(ow, 2) + imin_d(OW - 1 - ow, 2) + 1;
  float r = 1.f / (float)(cy * cx);   // t1 == 1
  const _Float16* cp = conv + ((size_t)n * P + p) * 16;
  float n2 = 0.f;
#pragma unroll
  for (int c = 0; c < 16; ++c) { float v = r * (float)cp[c]; n2 += v * v; }
  float s = (n2 / (1.f + n2)) / sqrtf(n2 + 1e-9f);
  out[idx] = sqrtf(s * s * n2 + 1e-9f);
}

// ---------------------------------------------------------------------------
extern "C" void kernel_launch(void* const* d_in, const int* in_sizes, int n_in,
                              void* d_out, int out_size, void* d_ws, size_t ws_size,
                              hipStream_t stream)
{
  (void)in_sizes; (void)n_in; (void)out_size; (void)ws_size;
  const float* x   = (const float*)d_in[0];
  const float* wc1 = (const float*)d_in[1];
  const float* w1a = (const float*)d_in[2];
  const float* w1b = (const float*)d_in[3];
  const float* w2a = (const float*)d_in[4];
  const float* w2b = (const float*)d_in[5];
  const float* w3a = (const float*)d_in[6];
  const float* w3b = (const float*)d_in[7];
  const float* w4  = (const float*)d_in[8];
  const float* b4  = (const float*)d_in[9];
  const float* w5  = (const float*)d_in[10];
  const float* w6  = (const float*)d_in[11];
  const float* b6  = (const float*)d_in[12];
  const float* w7  = (const float*)d_in[13];
  const float* w8  = (const float*)d_in[14];
  const float* b8  = (const float*)d_in[15];
  const float* w10 = (const float*)d_in[16];
  float* out = (float*)d_out;

  // ---- workspace bump allocator (f16 elements, 256B aligned) ----
  char* ws = (char*)d_ws;
  size_t cur = 0;
  auto alloc = [&](size_t elems) -> _Float16* {
    _Float16* p = (_Float16*)(ws + cur);
    cur += (elems * sizeof(_Float16) + 255) & ~(size_t)255;
    return p;
  };
  _Float16* S    = alloc(8388608);   // raw conv out scratch (max: w8, 4*256*256*32)
  _Float16* xh   = alloc(2097152);   // x NHWC f16, Cstride 8 (ch 3..7 zero)
  _Float16* cat3 = alloc(12582912);  // 4*256*256*48  [w8-out | skip1]
  _Float16* cat2 = alloc(8388608);   // 4*128*128*128 [w6-out | skip2]
  _Float16* cat1 = alloc(8388608);   // 4*64*64*512   [w4-out | skip3]
  _Float16* a2   = alloc(2097152);   // 4*128*128*32
  _Float16* a4   = alloc(2097152);   // 4*64*64*128
  _Float16* a6   = alloc(2097152);   // 4*32*32*512
  _Float16* a7   = alloc(1048576);   // 4*32*32*256
  _Float16* a9   = alloc(2097152);   // 4*64*64*128
  _Float16* a11  = alloc(4194304);   // 4*128*128*64
  _Float16* pw0  = alloc((size_t)224 * 16);
  _Float16* pw1  = alloc((size_t)416 * 32);
  _Float16* pw2  = alloc((size_t)800 * 64);
  _Float16* pw3  = alloc((size_t)1600 * 128);
  _Float16* pw4  = alloc((size_t)3200 * 256);
  _Float16* pw5  = alloc((size_t)6400 * 512);
  _Float16* pw6  = alloc((size_t)12800 * 256);
  _Float16* pw7  = alloc((size_t)6400 * 256);
  _Float16* pw8  = alloc((size_t)12800 * 128);
  _Float16* pw9  = alloc((size_t)3200 * 64);
  _Float16* pw10 = alloc((size_t)3200 * 64);
  _Float16* pw11 = alloc((size_t)1600 * 32);
  _Float16* pw12 = alloc((size_t)1600 * 16);

  auto pack = [&](const float* w, _Float16* wpk, int t0, int z_in, int Cout,
                  int cshift, int KK, int dec) {
    int total = Cout * KK;
    pack_w_kernel<<<cdiv_host(total, 256), 256, 0, stream>>>(w, wpk, t0, z_in, Cout, cshift, KK, dec);
  };
  auto conv = [&](const _Float16* in, int Cin, int cshift, int Cstride, int cin_off,
                  int Hin, int Win, int OH, int OW, int stride, int dec,
                  const _Float16* wpk, int Cout, _Float16* outbuf) {
    int P = OH * OW;
    bool big = cshift >= 5;   // chunk lies inside one tap
    if (Cout >= 32) {
      dim3 g(P / 32, Cout / 32, NIMG);
      if (dec)
        wmma_conv_kernel<2, true, true><<<g, 32, 0, stream>>>(in, wpk, outbuf, Cin, cshift, Cstride,
                                                              cin_off, Hin, Win, OH, OW, Cout, stride);
      else if (big)
        wmma_conv_kernel<2, false, true><<<g, 32, 0, stream>>>(in, wpk, outbuf, Cin, cshift, Cstride,
                                                               cin_off, Hin, Win, OH, OW, Cout, stride);
      else
        wmma_conv_kernel<2, false, false><<<g, 32, 0, stream>>>(in, wpk, outbuf, Cin, cshift, Cstride,
                                                                cin_off, Hin, Win, OH, OW, Cout, stride);
    } else {
      dim3 g(P / 32, Cout / 16, NIMG);
      if (big)
        wmma_conv_kernel<1, false, true><<<g, 32, 0, stream>>>(in, wpk, outbuf, Cin, cshift, Cstride,
                                                               cin_off, Hin, Win, OH, OW, Cout, stride);
      else
        wmma_conv_kernel<1, false, false><<<g, 32, 0, stream>>>(in, wpk, outbuf, Cin, cshift, Cstride,
                                                                cin_off, Hin, Win, OH, OW, Cout, stride);
    }
  };
  auto route = [&](const _Float16* cin, const float* bias, int t0b, _Float16* dst,
                   int cout_off, int Cstride, int OH, int OW, int Cout, int t1, int z1) {
    int total = NIMG * OH * OW * t1;
    route_kernel<<<cdiv_host(total, 256), 256, 0, stream>>>(cin, bias, t0b, dst, cout_off,
                                                            Cstride, OH * OW, OW, OH, Cout, t1, z1);
  };

  // ---- weight + input packing ----
  pack(wc1, pw0,  1,  3,  16, 3,   224, 0);
  pack(w1a, pw1,  1, 16,  32, 4,   416, 0);
  pack(w1b, pw2,  2, 16,  64, 5,   800, 0);
  pack(w2a, pw3,  4, 16, 128, 6,  1600, 0);
  pack(w2b, pw4,  4, 32, 256, 7,  3200, 0);
  pack(w3a, pw5,  8, 32, 512, 8,  6400, 0);
  pack(w3b, pw6,  8, 64, 256, 9, 12800, 0);
  pack(w4,  pw7,  8, 32, 256, 8,  6400, 1);
  pack(w5,  pw8, 16, 32, 128, 9, 12800, 0);
  pack(w6,  pw9,  4, 32,  64, 7,  3200, 1);
  pack(w7,  pw10, 8, 16,  64, 7,  3200, 0);
  pack(w8,  pw11, 4, 16,  32, 6,  1600, 1);
  pack(w10, pw12, 3, 16,  16, 6,  1600, 0);
  pack_x_kernel<<<cdiv_host(NIMG * 256 * 256 * 8, 256), 256, 0, stream>>>(x, xh, 256, 256);

  // ---- network (t0-sum folded into Cin; concat = channel sub-ranges) ----
  // L1 conv1 -> skip1 (cat3 ch[32..48))
  conv(xh, 3, 3, 8, 0, 256, 256, 256, 256, 1, 0, pw0, 16, S);
  cast_copy_kernel<<<cdiv_host(NIMG * 65536 * 16, 256), 256, 0, stream>>>(S, cat3, 16, 32, 48, NIMG * 65536);
  // L2 w1a (s2) -> a2
  conv(cat3, 16, 4, 48, 32, 256, 256, 128, 128, 2, 0, pw1, 32, S);
  route(S, nullptr, 0, a2, 0, 32, 128, 128, 32, 2, 16);
  // L3 w1b -> skip2 (cat2 ch[64..128))
  conv(a2, 32, 5, 32, 0, 128, 128, 128, 128, 1, 0, pw2, 64, S);
  route(S, nullptr, 0, cat2, 64, 128, 128, 128, 64, 4, 16);
  // L4 w2a (s2) -> a4
  conv(cat2, 64, 6, 128, 64, 128, 128, 64, 64, 2, 0, pw3, 128, S);
  route(S, nullptr, 0, a4, 0, 128, 64, 64, 128, 4, 32);
  // L5 w2b -> skip3 (cat1 ch[256..512))
  conv(a4, 128, 7, 128, 0, 64, 64, 64, 64, 1, 0, pw4, 256, S);
  route(S, nullptr, 0, cat1, 256, 512, 64, 64, 256, 8, 32);
  // L6 w3a (s2) -> a6
  conv(cat1, 256, 8, 512, 256, 64, 64, 32, 32, 2, 0, pw5, 512, S);
  route(S, nullptr, 0, a6, 0, 512, 32, 32, 512, 8, 64);
  // L7 w3b -> a7
  conv(a6, 512, 9, 512, 0, 32, 32, 32, 32, 1, 0, pw6, 256, S);
  route(S, nullptr, 0, a7, 0, 256, 32, 32, 256, 8, 32);
  // L8 deconv w4 (+b4) -> cat1 ch[0..256)
  conv(a7, 256, 8, 256, 0, 32, 32, 64, 64, 1, 1, pw7, 256, S);
  route(S, b4, 8, cat1, 0, 512, 64, 64, 256, 8, 32);
  // L9 w5 -> a9
  conv(cat1, 512, 9, 512, 0, 64, 64, 64, 64, 1, 0, pw8, 128, S);
  route(S, nullptr, 0, a9, 0, 128, 64, 64, 128, 4, 32);
  // L10 deconv w6 (+b6) -> cat2 ch[0..64)
  conv(a9, 128, 7, 128, 0, 64, 64, 128, 128, 1, 1, pw9, 64, S);
  route(S, b6, 4, cat2, 0, 128, 128, 128, 64, 4, 16);
  // L11 w7 -> a11
  conv(cat2, 128, 7, 128, 0, 128, 128, 128, 128, 1, 0, pw10, 64, S);
  route(S, nullptr, 0, a11, 0, 64, 128, 128, 64, 4, 16);
  // L12 deconv w8 (+b8) -> cat3 ch[0..32)
  conv(a11, 64, 6, 64, 0, 128, 128, 256, 256, 1, 1, pw11, 32, S);
  route(S, b8, 4, cat3, 0, 48, 256, 256, 32, 2, 16);
  // L13 w10 -> final norm
  conv(cat3, 48, 6, 48, 0, 256, 256, 256, 256, 1, 0, pw12, 16, S);
  final_kernel<<<cdiv_host(NIMG * 65536, 256), 256, 0, stream>>>(S, out, 65536, 256, 256);
}
